// ListenerModelBertAttCtxHist_87247965651405
// MI455X (gfx1250) — compile-verified, matrix operand order checked by reading, not run on hardware
//
#include <hip/hip_runtime.h>
#include <math.h>
#include <stdint.h>

typedef __bf16 bf16;
typedef __attribute__((ext_vector_type(16))) __bf16 v16bf;
typedef __attribute__((ext_vector_type(8)))  __bf16 v8bf;
typedef __attribute__((ext_vector_type(8)))  float  v8f;

#define B_  32
#define L_  512
#define S_  6
#define HL_ 128
#define E_  768
#define HID_ 512
#define IMG_ 2048
#define ATT_ 256
#define BL_ (B_*L_)
#define BS_ (B_*S_)

#define LDS_STRIDE 40   // bf16 elements per LDS tile row (32 data + 8 pad -> conflict-free b128)

// ---------------- block reductions (wave32) ----------------
__device__ __forceinline__ float block_sum(float v, float* sm) {
#pragma unroll
  for (int off = 16; off > 0; off >>= 1) v += __shfl_down(v, off, 32);
  int lane = threadIdx.x & 31, wid = threadIdx.x >> 5;
  int nw = (int)(blockDim.x >> 5);
  if (lane == 0) sm[wid] = v;
  __syncthreads();
  float r = ((int)threadIdx.x < nw) ? sm[threadIdx.x] : 0.f;
  if (wid == 0) {
#pragma unroll
    for (int off = 16; off > 0; off >>= 1) r += __shfl_down(r, off, 32);
    if (lane == 0) sm[0] = r;
  }
  __syncthreads();
  r = sm[0];
  __syncthreads();
  return r;
}

__device__ __forceinline__ float block_max(float v, float* sm) {
#pragma unroll
  for (int off = 16; off > 0; off >>= 1) v = fmaxf(v, __shfl_down(v, off, 32));
  int lane = threadIdx.x & 31, wid = threadIdx.x >> 5;
  int nw = (int)(blockDim.x >> 5);
  if (lane == 0) sm[wid] = v;
  __syncthreads();
  float r = ((int)threadIdx.x < nw) ? sm[threadIdx.x] : -INFINITY;
  if (wid == 0) {
#pragma unroll
    for (int off = 16; off > 0; off >>= 1) r = fmaxf(r, __shfl_down(r, off, 32));
    if (lane == 0) sm[0] = r;
  }
  __syncthreads();
  r = sm[0];
  __syncthreads();
  return r;
}

// ---------------- weight prep ----------------
__global__ void transpose_to_bf16(const float* __restrict__ src, bf16* __restrict__ dst,
                                  int K, int N, int koff) {
  int idx = blockIdx.x * blockDim.x + threadIdx.x;
  if (idx >= K * N) return;
  int n = idx / K, k = idx - n * K;
  dst[(size_t)n * K + k] = (bf16)src[(size_t)(k + koff) * N + n];
}

__global__ void convert_f32_to_bf16(const float* __restrict__ src, bf16* __restrict__ dst, int n) {
  int idx = blockIdx.x * blockDim.x + threadIdx.x;
  if (idx < n) dst[idx] = (bf16)src[idx];
}

// ---------------- fused embed + LayerNorm (text) ----------------
__global__ void embed_ln_text(const int* __restrict__ ids, const float* __restrict__ we,
                              const float* __restrict__ pe, const float* __restrict__ te,
                              const float* __restrict__ g, const float* __restrict__ bta,
                              bf16* __restrict__ out) {
  __shared__ float sm[32];
  int r = blockIdx.x;
  int l = r & (L_ - 1);
  int id = ids[r];
  float x[3];
  float s = 0.f;
#pragma unroll
  for (int i = 0; i < 3; ++i) {
    int e = threadIdx.x + i * 256;
    x[i] = we[(size_t)id * E_ + e] + pe[(size_t)l * E_ + e] + te[e];
    s += x[i];
  }
  float mu = block_sum(s, sm) * (1.f / (float)E_);
  float q = 0.f;
#pragma unroll
  for (int i = 0; i < 3; ++i) { float d = x[i] - mu; q += d * d; }
  float var = block_sum(q, sm) * (1.f / (float)E_);
  float rstd = rsqrtf(var + 1e-12f);
#pragma unroll
  for (int i = 0; i < 3; ++i) {
    int e = threadIdx.x + i * 256;
    out[(size_t)r * E_ + e] = (bf16)(g[e] * (x[i] - mu) * rstd + bta[e]);
  }
}

// ---------------- fused embed + LN + masked average (history) ----------------
__global__ void embed_ln_hist_avg(const int* __restrict__ hist, const int* __restrict__ hlen,
                                  const float* __restrict__ we, const float* __restrict__ pe,
                                  const float* __restrict__ te, const float* __restrict__ g,
                                  const float* __restrict__ bta, bf16* __restrict__ out) {
  __shared__ float sm[32];
  int bs = blockIdx.x;
  int len = hlen[bs];
  float acc[3] = {0.f, 0.f, 0.f};
  for (int h = 0; h < len; ++h) {
    int id = hist[(size_t)bs * HL_ + h];
    float x[3];
    float s = 0.f;
#pragma unroll
    for (int i = 0; i < 3; ++i) {
      int e = threadIdx.x + i * 256;
      x[i] = we[(size_t)id * E_ + e] + pe[(size_t)h * E_ + e] + te[e];
      s += x[i];
    }
    float mu = block_sum(s, sm) * (1.f / (float)E_);
    float q = 0.f;
#pragma unroll
    for (int i = 0; i < 3; ++i) { float d = x[i] - mu; q += d * d; }
    float var = block_sum(q, sm) * (1.f / (float)E_);
    float rstd = rsqrtf(var + 1e-12f);
#pragma unroll
    for (int i = 0; i < 3; ++i) {
      int e = threadIdx.x + i * 256;
      acc[i] += g[e] * (x[i] - mu) * rstd + bta[e];
    }
  }
  float inv = 1.f / (float)(len > 0 ? len : 1);
#pragma unroll
  for (int i = 0; i < 3; ++i) {
    int e = threadIdx.x + i * 256;
    out[(size_t)bs * E_ + e] = (bf16)(acc[i] * inv);
  }
}

// ---------------- LDS-staged async double-buffered WMMA GEMM ----------------
// C[M,N] = act(A[M,K] @ Bt[N,K]^T + bias + rowbias)
// 256 threads = 8 waves (2 M x 4 N), block tile 128x128, K step 32.
__global__ __launch_bounds__(256) void wmma_gemm_bf16(
    const bf16* __restrict__ A, const bf16* __restrict__ Bt,
    const float* __restrict__ bias, const float* __restrict__ rowbias, int rowdiv,
    float* __restrict__ outf, bf16* __restrict__ outb,
    int M, int N, int K, int act) {
  __shared__ __align__(16) bf16 sA[2][128 * LDS_STRIDE];
  __shared__ __align__(16) bf16 sB[2][128 * LDS_STRIDE];

  int tid  = threadIdx.x;
  int lane = tid & 31;
  int wid  = tid >> 5;
  int wm = wid & 1;          // 2 waves over M
  int wn = wid >> 1;         // 4 waves over N
  int mb_blk = blockIdx.y * 128;
  int nb_blk = blockIdx.x * 128;
  int mrow  = lane & 15;
  int khalf = (lane >> 4) << 3;  // 0 or 8

  // cooperative async global->LDS stage of one 128x32 A tile + 128x32 B tile
  auto stage = [&](int buf, int k0) {
#pragma unroll
    for (int q = 0; q < 2; ++q) {
      int idx = q * 256 + tid;       // 0..511
      int row = idx >> 2;            // 0..127
      int seg = idx & 3;             // 16B segment within 64B row
      int ar = mb_blk + row; if (ar > M - 1) ar = M - 1;
      unsigned la = (unsigned)(uintptr_t)&sA[buf][row * LDS_STRIDE + seg * 8];
      unsigned long long ga =
          (unsigned long long)(uintptr_t)(A + (size_t)ar * K + k0 + seg * 8);
      asm volatile("global_load_async_to_lds_b128 %0, %1, off"
                   :: "v"(la), "v"(ga) : "memory");
      int br = nb_blk + row;
      unsigned lb = (unsigned)(uintptr_t)&sB[buf][row * LDS_STRIDE + seg * 8];
      unsigned long long gb =
          (unsigned long long)(uintptr_t)(Bt + (size_t)br * K + k0 + seg * 8);
      asm volatile("global_load_async_to_lds_b128 %0, %1, off"
                   :: "v"(lb), "v"(gb) : "memory");
    }
    if (k0 + 64 < K)  // keep the gfx1250 prefetch path warm two tiles ahead
      __builtin_prefetch((const void*)(A + (size_t)(mb_blk + mrow) * K + k0 + 64), 0, 1);
  };

  v8f acc[4][2];
#pragma unroll
  for (int i = 0; i < 4; ++i)
#pragma unroll
    for (int j = 0; j < 2; ++j)
#pragma unroll
      for (int e = 0; e < 8; ++e) acc[i][j][e] = 0.f;

  int T = K >> 5;
  stage(0, 0);
  for (int t = 0; t < T; ++t) {
    int buf = t & 1;
    if (t + 1 < T) {
      stage(buf ^ 1, (t + 1) << 5);
      // async loads complete in order: <=4 outstanding => tile t is resident
      asm volatile("s_wait_asynccnt 0x4" ::: "memory");
    } else {
      asm volatile("s_wait_asynccnt 0x0" ::: "memory");
    }
    __syncthreads();

    v16bf af[4];
#pragma unroll
    for (int i = 0; i < 4; ++i) {
      const bf16* p = &sA[buf][(wm * 64 + 16 * i + mrow) * LDS_STRIDE + khalf];
      v8bf lo = *(const v8bf*)p;
      v8bf hi = *(const v8bf*)(p + 16);
#pragma unroll
      for (int e = 0; e < 8; ++e) { af[i][e] = lo[e]; af[i][e + 8] = hi[e]; }
    }
    v16bf bfv[2];
#pragma unroll
    for (int j = 0; j < 2; ++j) {
      const bf16* p = &sB[buf][(wn * 32 + 16 * j + mrow) * LDS_STRIDE + khalf];
      v8bf lo = *(const v8bf*)p;
      v8bf hi = *(const v8bf*)(p + 16);
#pragma unroll
      for (int e = 0; e < 8; ++e) { bfv[j][e] = lo[e]; bfv[j][e + 8] = hi[e]; }
    }
#pragma unroll
    for (int i = 0; i < 4; ++i)
#pragma unroll
      for (int j = 0; j < 2; ++j)
        acc[i][j] = __builtin_amdgcn_wmma_f32_16x16x32_bf16(
            false, af[i], false, bfv[j], (short)0, acc[i][j], false, false);
    __syncthreads();  // protect buffer about to be overwritten next iteration
  }

  // epilogue (C/D layout: vgpr r -> M = r + 8*(lane>>4), N = lane&15)
  int mb = mb_blk + wm * 64;
  int nb = nb_blk + wn * 32;
#pragma unroll
  for (int i = 0; i < 4; ++i) {
#pragma unroll
    for (int j = 0; j < 2; ++j) {
      int n = nb + 16 * j + mrow;
      float bv = bias ? bias[n] : 0.f;
#pragma unroll
      for (int r = 0; r < 8; ++r) {
        int m = mb + 16 * i + r + khalf;
        float v = acc[i][j][r] + bv;
        if (rowdiv) v += rowbias[(size_t)(m / rowdiv) * N + n];
        if (act == 1) v = fmaxf(v, 0.f);
        else if (act == 2) v = tanhf(v);
        if (m < M) {
          size_t o = (size_t)m * N + n;
          if (outb) outb[o] = (bf16)v;
          else outf[o] = v;
        }
      }
    }
  }
}

// ---------------- attention score ----------------
__global__ void att_score(const bf16* __restrict__ t, const float* __restrict__ wa2,
                          const float* __restrict__ ba2, const unsigned char* __restrict__ masks,
                          float* __restrict__ att) {
  __shared__ float sm[32];
  int r = blockIdx.x;
  float v = (float)t[(size_t)r * ATT_ + threadIdx.x] * wa2[threadIdx.x];
  float s = block_sum(v, sm);
  if (threadIdx.x == 0) att[r] = masks[r] ? -INFINITY : (s + ba2[0]);
}

// ---------------- softmax over L per batch ----------------
__global__ void softmax_l(const float* __restrict__ att, float* __restrict__ w) {
  __shared__ float sm[32];
  int b = blockIdx.x, l = threadIdx.x;
  float v = att[b * L_ + l];
  float mx = block_max(v, sm);
  float e = expf(v - mx);
  float s = block_sum(e, sm);
  w[b * L_ + l] = e / s;
}

// ---------------- attended[b,n] = sum_l w[b,l]*mm[b,l,n] ----------------
__global__ void attend(const bf16* __restrict__ mm, const float* __restrict__ w,
                       float* __restrict__ out) {
  int b = blockIdx.x;
  int n = blockIdx.y * 128 + threadIdx.x;
  float acc = 0.f;
  for (int l = 0; l < L_; ++l)
    acc += w[b * L_ + l] * (float)mm[((size_t)(b * L_ + l)) * HID_ + n];
  out[b * HID_ + n] = acc;
}

// ---------------- sep combine ----------------
__global__ void sep_combine(const float* __restrict__ sepg, const float* __restrict__ contrib,
                            const int* __restrict__ hlen, float* __restrict__ out) {
  __shared__ float sm[32];
  int b = blockIdx.x, n = threadIdx.x;
  float acc = 0.f;
  for (int s = 0; s < S_; ++s) {
    int row = b * S_ + s;
    float v = sepg[(size_t)row * HID_ + n];
    if (hlen[row] > 0) v += contrib[(size_t)row * HID_ + n];
    v = fmaxf(v, 0.f);
    float ss = block_sum(v * v, sm);
    float nrm = sqrtf(ss);
    acc += v / fmaxf(nrm, 1e-12f);
  }
  out[b * HID_ + n] = acc * (1.f / (float)S_);
}

// ---------------- final head ----------------
__global__ void final_head(const float* __restrict__ sepm, const float* __restrict__ attn,
                           const float* __restrict__ Wf1, const float* __restrict__ bf1,
                           const float* __restrict__ Wf2, const float* __restrict__ bf2,
                           float* __restrict__ out) {
  __shared__ float hs[HID_];
  int b = blockIdx.x;
#pragma unroll
  for (int j = 0; j < 2; ++j) {
    int n = threadIdx.x + j * 256;
    float acc = bf1[n];
    for (int k = 0; k < HID_; ++k) acc += sepm[b * HID_ + k] * Wf1[(size_t)k * HID_ + n];
    for (int k = 0; k < HID_; ++k) acc += attn[b * HID_ + k] * Wf1[(size_t)(HID_ + k) * HID_ + n];
    hs[n] = 0.5f * acc * (1.f + erff(acc * 0.70710678118654752f));
  }
  __syncthreads();
  if (threadIdx.x < 2) {
    float acc = bf2[threadIdx.x];
    for (int k = 0; k < HID_; ++k) acc += hs[k] * Wf2[k * 2 + threadIdx.x];
    out[b * 2 + threadIdx.x] = acc;
  }
}

// ---------------- host launcher ----------------
extern "C" void kernel_launch(void* const* d_in, const int* in_sizes, int n_in,
                              void* d_out, int out_size, void* d_ws, size_t ws_size,
                              hipStream_t stream) {
  const float* separate_images = (const float*)d_in[0];
  const float* img_pred  = (const float*)d_in[1];
  const int*   input_text = (const int*)d_in[2];
  const int*   prev_hist  = (const int*)d_in[3];
  const int*   prev_hist_len = (const int*)d_in[4];
  const unsigned char* masks = (const unsigned char*)d_in[5];
  const float* word_emb = (const float*)d_in[6];
  const float* pos_emb  = (const float*)d_in[7];
  const float* type_emb = (const float*)d_in[8];
  const float* ln_g = (const float*)d_in[9];
  const float* ln_b = (const float*)d_in[10];
  const float* W_sep = (const float*)d_in[11]; const float* b_sep = (const float*)d_in[12];
  const float* W_e2h = (const float*)d_in[13]; const float* b_e2h = (const float*)d_in[14];
  const float* W_hist = (const float*)d_in[15]; const float* b_hist = (const float*)d_in[16];
  const float* W_img = (const float*)d_in[17]; const float* b_img = (const float*)d_in[18];
  const float* W_mm  = (const float*)d_in[19]; const float* b_mm  = (const float*)d_in[20];
  const float* W_a1  = (const float*)d_in[21]; const float* b_a1  = (const float*)d_in[22];
  const float* W_a2  = (const float*)d_in[23]; const float* b_a2  = (const float*)d_in[24];
  const float* W_f1  = (const float*)d_in[25]; const float* b_f1  = (const float*)d_in[26];
  const float* W_f2  = (const float*)d_in[27]; const float* b_f2  = (const float*)d_in[28];
  float* out = (float*)d_out;

  char* ws = (char*)d_ws;
  size_t off = 0;
  auto take = [&](size_t bytes) -> void* {
    void* p = ws + off;
    off += (bytes + 255) & ~(size_t)255;
    return p;
  };

  bf16* reps_bf = (bf16*)take((size_t)BL_ * E_ * 2);
  bf16* ir_bf   = (bf16*)take((size_t)BL_ * HID_ * 2);
  bf16* mm_bf   = (bf16*)take((size_t)BL_ * HID_ * 2);
  bf16* t_bf    = (bf16*)take((size_t)BL_ * ATT_ * 2);
  bf16* img_bf  = (bf16*)take((size_t)B_ * IMG_ * 2);
  bf16* simg_bf = (bf16*)take((size_t)BS_ * IMG_ * 2);
  bf16* havg_bf = (bf16*)take((size_t)BS_ * E_ * 2);
  bf16* proj_bf = (bf16*)take((size_t)B_ * HID_ * 2);
  bf16* wt_e2h  = (bf16*)take((size_t)HID_ * E_ * 2);
  bf16* wt_img  = (bf16*)take((size_t)HID_ * IMG_ * 2);
  bf16* wt_mmt  = (bf16*)take((size_t)HID_ * HID_ * 2);
  bf16* wt_mmb  = (bf16*)take((size_t)HID_ * HID_ * 2);
  bf16* wt_a1   = (bf16*)take((size_t)ATT_ * HID_ * 2);
  bf16* wt_hist = (bf16*)take((size_t)HID_ * E_ * 2);
  bf16* wt_sep  = (bf16*)take((size_t)HID_ * IMG_ * 2);
  float* immm_f   = (float*)take((size_t)B_ * HID_ * 4);
  float* att_f    = (float*)take((size_t)BL_ * 4);
  float* w_f      = (float*)take((size_t)BL_ * 4);
  float* attn_f   = (float*)take((size_t)B_ * HID_ * 4);
  float* contrib_f= (float*)take((size_t)BS_ * HID_ * 4);
  float* sepg_f   = (float*)take((size_t)BS_ * HID_ * 4);
  float* sepm_f   = (float*)take((size_t)B_ * HID_ * 4);
  (void)ws_size; (void)n_in; (void)in_sizes; (void)out_size;

  // ---- weight prep ----
  auto tgrid = [](int k, int n) { return dim3((k * n + 255) / 256); };
  transpose_to_bf16<<<tgrid(E_, HID_), 256, 0, stream>>>(W_e2h, wt_e2h, E_, HID_, 0);
  transpose_to_bf16<<<tgrid(IMG_, HID_), 256, 0, stream>>>(W_img, wt_img, IMG_, HID_, 0);
  transpose_to_bf16<<<tgrid(HID_, HID_), 256, 0, stream>>>(W_mm, wt_mmt, HID_, HID_, 0);
  transpose_to_bf16<<<tgrid(HID_, HID_), 256, 0, stream>>>(W_mm, wt_mmb, HID_, HID_, HID_);
  transpose_to_bf16<<<tgrid(HID_, ATT_), 256, 0, stream>>>(W_a1, wt_a1, HID_, ATT_, 0);
  transpose_to_bf16<<<tgrid(E_, HID_), 256, 0, stream>>>(W_hist, wt_hist, E_, HID_, 0);
  transpose_to_bf16<<<tgrid(IMG_, HID_), 256, 0, stream>>>(W_sep, wt_sep, IMG_, HID_, 0);
  convert_f32_to_bf16<<<(B_ * IMG_ + 255) / 256, 256, 0, stream>>>(img_pred, img_bf, B_ * IMG_);
  convert_f32_to_bf16<<<(BS_ * IMG_ + 255) / 256, 256, 0, stream>>>(separate_images, simg_bf, BS_ * IMG_);

  // ---- embeddings ----
  embed_ln_text<<<BL_, 256, 0, stream>>>(input_text, word_emb, pos_emb, type_emb, ln_g, ln_b, reps_bf);
  embed_ln_hist_avg<<<BS_, 256, 0, stream>>>(prev_hist, prev_hist_len, word_emb, pos_emb, type_emb,
                                             ln_g, ln_b, havg_bf);

  // ---- WMMA GEMMs (block 256 threads, tile 128x128) ----
  dim3 blk(256);
  wmma_gemm_bf16<<<dim3(HID_ / 128, BL_ / 128), blk, 0, stream>>>(
      reps_bf, wt_e2h, b_e2h, nullptr, 0, nullptr, ir_bf, BL_, HID_, E_, 1);
  wmma_gemm_bf16<<<dim3(HID_ / 128, 1), blk, 0, stream>>>(
      img_bf, wt_img, b_img, nullptr, 0, nullptr, proj_bf, B_, HID_, IMG_, 1);
  wmma_gemm_bf16<<<dim3(HID_ / 128, 1), blk, 0, stream>>>(
      proj_bf, wt_mmb, b_mm, nullptr, 0, immm_f, nullptr, B_, HID_, HID_, 0);
  wmma_gemm_bf16<<<dim3(HID_ / 128, BL_ / 128), blk, 0, stream>>>(
      ir_bf, wt_mmt, nullptr, immm_f, L_, nullptr, mm_bf, BL_, HID_, HID_, 1);
  wmma_gemm_bf16<<<dim3(ATT_ / 128, BL_ / 128), blk, 0, stream>>>(
      mm_bf, wt_a1, b_a1, nullptr, 0, nullptr, t_bf, BL_, ATT_, HID_, 2);
  wmma_gemm_bf16<<<dim3(HID_ / 128, (BS_ + 127) / 128), blk, 0, stream>>>(
      simg_bf, wt_sep, b_sep, nullptr, 0, sepg_f, nullptr, BS_, HID_, IMG_, 0);
  wmma_gemm_bf16<<<dim3(HID_ / 128, (BS_ + 127) / 128), blk, 0, stream>>>(
      havg_bf, wt_hist, b_hist, nullptr, 0, contrib_f, nullptr, BS_, HID_, E_, 1);

  // ---- attention pooling ----
  att_score<<<BL_, ATT_, 0, stream>>>(t_bf, W_a2, b_a2, masks, att_f);
  softmax_l<<<B_, L_, 0, stream>>>(att_f, w_f);
  attend<<<dim3(B_, HID_ / 128), 128, 0, stream>>>(mm_bf, w_f, attn_f);

  // ---- image/history combine ----
  sep_combine<<<B_, HID_, 0, stream>>>(sepg_f, contrib_f, prev_hist_len, sepm_f);

  // ---- head ----
  final_head<<<B_, 256, 0, stream>>>(sepm_f, attn_f, W_f1, b_f1, W_f2, b_f2, out);
}
